// RecurrentNetwork_38036230373607
// MI455X (gfx1250) — matrix operand order repaired and checked by reading, not compiled, not run on hardware
//
#include <hip/hip_runtime.h>

// Problem constants
#define BB  128   // batch
#define HH  512   // hidden
#define TT  512   // timesteps
#define FF  64    // input features (layer 0)
#define NWG 32    // persistent workgroups; each owns HC h-columns
#define HC  16

typedef _Float16 h8v  __attribute__((ext_vector_type(8)));
typedef _Float16 v16h __attribute__((ext_vector_type(16)));
typedef float    v8f  __attribute__((ext_vector_type(8)));
typedef float    f8v  __attribute__((ext_vector_type(8)));

__device__ __forceinline__ float sigmoidf_(float x) {
  return 1.0f / (1.0f + __expf(-x));
}

// ---------------- prep kernels ----------------
__global__ void cvt_f16_kernel(const float* __restrict__ s, _Float16* __restrict__ d, int n) {
  int i = blockIdx.x * blockDim.x + threadIdx.x;
  if (i < n) d[i] = (_Float16)s[i];
}

__global__ void bias_combine_kernel(const float* __restrict__ a, const float* __restrict__ b,
                                    float* __restrict__ o, int n) {
  int i = blockIdx.x * blockDim.x + threadIdx.x;
  if (i < n) o[i] = a[i] + b[i];
}

__global__ void init_state_kernel(_Float16* hp0, _Float16* hp1, unsigned* bars) {
  int i = blockIdx.x * blockDim.x + threadIdx.x;
  if (i < BB * HH) { hp0[i] = (_Float16)0.0f; hp1[i] = (_Float16)0.0f; }
  if (i < 64) bars[i] = 0u;
}

// ---------------- persistent LSTM layer kernel ----------------
// Grid: NWG blocks x 256 threads (8 waves of 32). Block wg owns h-columns
// [wg*HC, wg*HC+16) i.e. gate rows {g*512 + wg*16 + jj : g in 0..3}.
// Per step: gates(128x64) = h_prev(128x512) @ w_hh_slice^T  +  x_t @ w_ih_slice^T
// via v_wmma_f32_16x16x32_f16; cell update elementwise; grid barrier.
template<int KIN, bool XF32, bool WRITE_SEQ, bool WRITE_LAST>
__global__ __launch_bounds__(256)
void lstm_layer_kernel(const void* __restrict__ xin,          // layer0: f32 x (B,T,F); layer1: f16 (T,B,H)
                       const _Float16* __restrict__ w_ih,     // (4H, KIN) f16
                       const _Float16* __restrict__ w_hh,     // (4H, H) f16
                       const float* __restrict__ bias,        // (4H,) combined b_ih+b_hh
                       _Float16* __restrict__ h_ping,         // (B,H) f16, zero-initialized
                       _Float16* __restrict__ h_pong,         // (B,H) f16
                       _Float16* __restrict__ h_seq,          // (T,B,H) f16 output sequence (WRITE_SEQ)
                       float*    __restrict__ h_last,         // (B,H) f32 final h (WRITE_LAST)
                       unsigned* __restrict__ bar)            // [0]=count, [1]=generation
{
  __shared__ _Float16 whh_lds[64 * HH];     // 64 KB: 4 gates x 16 cols, K=512
  __shared__ _Float16 wih_lds[64 * KIN];    // 8 KB (L0) / 64 KB (L1)
  __shared__ float    gates_lds[BB * 64];   // 32 KB pre-activations
  __shared__ float    bias_lds[64];

  const int tid  = threadIdx.x;
  const int wg   = blockIdx.x;
  const int j0   = wg * HC;
  const int wave = tid >> 5;
  const int lane = tid & 31;
  const int lrow = lane & 15;   // matrix row (A) / column (B) within tile
  const int lhi  = lane >> 4;   // K-half selector per ISA 16-bit layouts
  const int m0   = wave * 16;   // M-tile base (batch rows)
  const int arow = m0 + lrow;

  // Stage w_hh slice into LDS: local row r = g*16+jj -> global row g*512 + j0 + jj
  for (int i = tid; i < 64 * (HH / 8); i += 256) {
    int r  = i / (HH / 8);
    int kb = (i % (HH / 8)) * 8;
    int gr = (r >> 4) * HH + j0 + (r & 15);
    *(h8v*)&whh_lds[r * HH + kb] = *(const h8v*)&w_hh[gr * HH + kb];
  }
  // Stage w_ih slice into LDS
  for (int i = tid; i < 64 * (KIN / 8); i += 256) {
    int r  = i / (KIN / 8);
    int kb = (i % (KIN / 8)) * 8;
    int gr = (r >> 4) * HH + j0 + (r & 15);
    *(h8v*)&wih_lds[r * KIN + kb] = *(const h8v*)&w_ih[gr * KIN + kb];
  }
  if (tid < 64) {
    int n = (tid >> 4) * HH + j0 + (tid & 15);
    bias_lds[tid] = bias[n];
  }
  __syncthreads();

  // Per-thread cell state: thread covers 8 of the 128x16 (b,jj) elements.
  float c[8];
#pragma unroll
  for (int e = 0; e < 8; ++e) c[e] = 0.0f;

  for (int t = 0; t < TT; ++t) {
    const _Float16* hprev = (t & 1) ? h_pong : h_ping;
    _Float16*       hnext = (t & 1) ? h_ping : h_pong;

    v8f acc[4];
#pragma unroll
    for (int g = 0; g < 4; ++g) acc[g] = (v8f){0.f,0.f,0.f,0.f,0.f,0.f,0.f,0.f};

    // ---- recurrent GEMM: K over h_prev (K = HH) ----
    for (int kb = 0; kb < HH; kb += 32) {
      // A fragment: 16x32 f16, row = arow; two contiguous 16B runs per lane
      h8v r0 = *(const h8v*)&hprev[arow * HH + kb + lhi * 8];
      h8v r1 = *(const h8v*)&hprev[arow * HH + kb + 16 + lhi * 8];
      v16h a;
#pragma unroll
      for (int i = 0; i < 8; ++i) { a[i] = r0[i]; a[8 + i] = r1[i]; }
#pragma unroll
      for (int g = 0; g < 4; ++g) {
        // B fragment: 32x16 f16, column = lrow; two contiguous 16B runs of w row
        h8v b0 = *(const h8v*)&whh_lds[(g * 16 + lrow) * HH + kb + lhi * 8];
        h8v b1 = *(const h8v*)&whh_lds[(g * 16 + lrow) * HH + kb + 16 + lhi * 8];
        v16h bf;
#pragma unroll
        for (int i = 0; i < 8; ++i) { bf[i] = b0[i]; bf[8 + i] = b1[i]; }
        acc[g] = __builtin_amdgcn_wmma_f32_16x16x32_f16(
            false, a, false, bf, (short)0, acc[g], false, false);
      }
    }

    // ---- input GEMM: K over x_t (K = KIN) ----
    for (int kb = 0; kb < KIN; kb += 32) {
      v16h a;
      if (XF32) {
        const float* x  = (const float*)xin;               // (B,T,F)
        const float* px = &x[(arow * TT + t) * KIN];
        f8v r0 = *(const f8v*)&px[kb + lhi * 8];
        f8v r1 = *(const f8v*)&px[kb + 16 + lhi * 8];
#pragma unroll
        for (int i = 0; i < 8; ++i) { a[i] = (_Float16)r0[i]; a[8 + i] = (_Float16)r1[i]; }
      } else {
        const _Float16* x  = (const _Float16*)xin;         // (T,B,H)
        const _Float16* px = &x[(t * BB + arow) * KIN];
        h8v r0 = *(const h8v*)&px[kb + lhi * 8];
        h8v r1 = *(const h8v*)&px[kb + 16 + lhi * 8];
#pragma unroll
        for (int i = 0; i < 8; ++i) { a[i] = r0[i]; a[8 + i] = r1[i]; }
      }
#pragma unroll
      for (int g = 0; g < 4; ++g) {
        h8v b0 = *(const h8v*)&wih_lds[(g * 16 + lrow) * KIN + kb + lhi * 8];
        h8v b1 = *(const h8v*)&wih_lds[(g * 16 + lrow) * KIN + kb + 16 + lhi * 8];
        v16h bf;
#pragma unroll
        for (int i = 0; i < 8; ++i) { bf[i] = b0[i]; bf[8 + i] = b1[i]; }
        acc[g] = __builtin_amdgcn_wmma_f32_16x16x32_f16(
            false, a, false, bf, (short)0, acc[g], false, false);
      }
    }

    // ---- dump accumulators to LDS (C layout: vgpr r -> row m0+r+8*lhi, col lrow) ----
#pragma unroll
    for (int g = 0; g < 4; ++g)
#pragma unroll
      for (int r = 0; r < 8; ++r)
        gates_lds[(m0 + r + 8 * lhi) * 64 + g * 16 + lrow] = acc[g][r];
    __syncthreads();

    // ---- elementwise LSTM cell: 2048 (b,jj) elements, 8 per thread ----
#pragma unroll
    for (int e = 0; e < 8; ++e) {
      int idx = tid * 8 + e;
      int b   = idx >> 4;
      int jj  = idx & 15;
      float pi = gates_lds[b * 64 +  0 + jj] + bias_lds[ 0 + jj];
      float pf = gates_lds[b * 64 + 16 + jj] + bias_lds[16 + jj];
      float pg = gates_lds[b * 64 + 32 + jj] + bias_lds[32 + jj];
      float po = gates_lds[b * 64 + 48 + jj] + bias_lds[48 + jj];
      float ig = sigmoidf_(pi);
      float fg = sigmoidf_(pf);
      float gg = tanhf(pg);
      float og = sigmoidf_(po);
      c[e] = fg * c[e] + ig * gg;
      float hv = og * tanhf(c[e]);
      hnext[b * HH + j0 + jj] = (_Float16)hv;
      if (WRITE_SEQ)  h_seq[(t * BB + b) * HH + j0 + jj] = (_Float16)hv;
      if (WRITE_LAST && t == TT - 1) h_last[b * HH + j0 + jj] = hv;
    }

    // ---- grid barrier (h_next fully written before any WG reads it) ----
    __threadfence();
    __syncthreads();
    if (tid == 0) {
      unsigned arrived = __hip_atomic_fetch_add(&bar[0], 1u, __ATOMIC_ACQ_REL,
                                                __HIP_MEMORY_SCOPE_AGENT);
      if (arrived == NWG - 1) {
        __hip_atomic_store(&bar[0], 0u, __ATOMIC_RELAXED, __HIP_MEMORY_SCOPE_AGENT);
        __hip_atomic_fetch_add(&bar[1], 1u, __ATOMIC_RELEASE, __HIP_MEMORY_SCOPE_AGENT);
      } else {
        while (__hip_atomic_load(&bar[1], __ATOMIC_ACQUIRE, __HIP_MEMORY_SCOPE_AGENT)
               < (unsigned)(t + 1)) {
          __builtin_amdgcn_s_sleep(1);
        }
      }
    }
    __syncthreads();
  }
}

// ---------------- final projection: out[b] = h_last[b,:] . lin_w + lin_b ----------------
__global__ void final_linear_kernel(const float* __restrict__ h_last,
                                    const float* __restrict__ lin_w,
                                    const float* __restrict__ lin_b,
                                    float* __restrict__ out) {
  __shared__ float red[256];
  int b = blockIdx.x, tid = threadIdx.x;
  float s = 0.0f;
  for (int j = tid; j < HH; j += 256) s += h_last[b * HH + j] * lin_w[j];
  red[tid] = s;
  __syncthreads();
  for (int w = 128; w > 0; w >>= 1) {
    if (tid < w) red[tid] += red[tid + w];
    __syncthreads();
  }
  if (tid == 0) out[b] = red[0] + lin_b[0];
}

// ---------------- workspace layout (bytes) ----------------
#define OFF_WIH0  ((size_t)0)          // 2048*64  f16 = 256 KB
#define OFF_WHH0  ((size_t)262144)     // 2048*512 f16 = 2 MB
#define OFF_WIH1  ((size_t)2359296)    // 2048*512 f16 = 2 MB
#define OFF_WHH1  ((size_t)4456448)    // 2048*512 f16 = 2 MB
#define OFF_BIAS0 ((size_t)6553600)    // 2048 f32
#define OFF_BIAS1 ((size_t)6561792)    // 2048 f32
#define OFF_HP0   ((size_t)6569984)    // 128*512 f16 ping (L0)
#define OFF_HQ0   ((size_t)6701056)    // 128*512 f16 pong (L0)
#define OFF_HP1   ((size_t)6832128)    // 128*512 f16 ping (L1)
#define OFF_HQ1   ((size_t)6963200)    // 128*512 f16 pong (L1)
#define OFF_HLAST ((size_t)7094272)    // 128*512 f32
#define OFF_BARS  ((size_t)7356416)    // 64 u32 (two cacheline-separated barriers)
#define OFF_H0SEQ ((size_t)7356672)    // 512*128*512 f16 = 64 MB

extern "C" void kernel_launch(void* const* d_in, const int* in_sizes, int n_in,
                              void* d_out, int out_size, void* d_ws, size_t ws_size,
                              hipStream_t stream) {
  const float* x      = (const float*)d_in[0];
  const float* w_ih_0 = (const float*)d_in[1];
  const float* w_hh_0 = (const float*)d_in[2];
  const float* b_ih_0 = (const float*)d_in[3];
  const float* b_hh_0 = (const float*)d_in[4];
  const float* w_ih_1 = (const float*)d_in[5];
  const float* w_hh_1 = (const float*)d_in[6];
  const float* b_ih_1 = (const float*)d_in[7];
  const float* b_hh_1 = (const float*)d_in[8];
  const float* lin_w  = (const float*)d_in[9];
  const float* lin_b  = (const float*)d_in[10];

  char* ws = (char*)d_ws;
  _Float16* wih0  = (_Float16*)(ws + OFF_WIH0);
  _Float16* whh0  = (_Float16*)(ws + OFF_WHH0);
  _Float16* wih1  = (_Float16*)(ws + OFF_WIH1);
  _Float16* whh1  = (_Float16*)(ws + OFF_WHH1);
  float*    bias0 = (float*)(ws + OFF_BIAS0);
  float*    bias1 = (float*)(ws + OFF_BIAS1);
  _Float16* hp0   = (_Float16*)(ws + OFF_HP0);
  _Float16* hq0   = (_Float16*)(ws + OFF_HQ0);
  _Float16* hp1   = (_Float16*)(ws + OFF_HP1);
  _Float16* hq1   = (_Float16*)(ws + OFF_HQ1);
  float*    hlast = (float*)(ws + OFF_HLAST);
  unsigned* bars  = (unsigned*)(ws + OFF_BARS);
  _Float16* h0seq = (_Float16*)(ws + OFF_H0SEQ);

  // Precision conversion + bias fold + state/barrier init (re-done every launch
  // so graph replay is deterministic).
  cvt_f16_kernel<<<(2048 * 64  + 255) / 256, 256, 0, stream>>>(w_ih_0, wih0, 2048 * 64);
  cvt_f16_kernel<<<(2048 * 512 + 255) / 256, 256, 0, stream>>>(w_hh_0, whh0, 2048 * 512);
  cvt_f16_kernel<<<(2048 * 512 + 255) / 256, 256, 0, stream>>>(w_ih_1, wih1, 2048 * 512);
  cvt_f16_kernel<<<(2048 * 512 + 255) / 256, 256, 0, stream>>>(w_hh_1, whh1, 2048 * 512);
  bias_combine_kernel<<<(2048 + 255) / 256, 256, 0, stream>>>(b_ih_0, b_hh_0, bias0, 2048);
  bias_combine_kernel<<<(2048 + 255) / 256, 256, 0, stream>>>(b_ih_1, b_hh_1, bias1, 2048);
  init_state_kernel<<<(BB * HH + 255) / 256, 256, 0, stream>>>(hp0, hp1, bars);

  // Layer 0: fused input GEMM (K=64, f32 x) + recurrence; writes (T,B,H) f16 sequence.
  lstm_layer_kernel<FF, true, true, false><<<NWG, 256, 0, stream>>>(
      (const void*)x, wih0, whh0, bias0, hp0, hq0, h0seq, nullptr, bars);

  // Layer 1: fused input GEMM (K=512, f16 sequence) + recurrence; writes final h (f32).
  lstm_layer_kernel<HH, false, false, true><<<NWG, 256, 0, stream>>>(
      (const void*)h0seq, wih1, whh1, bias1, hp1, hq1, nullptr, hlast, bars + 16);

  // out = h_last @ lin_w^T + lin_b  -> (128,1) f32
  final_linear_kernel<<<BB, 256, 0, stream>>>(hlast, lin_w, lin_b, (float*)d_out);
}
